// SVNSDEDeterministicVolatility_5935644803473
// MI455X (gfx1250) — matrix operand order repaired and checked by reading, not compiled, not run on hardware
//
#include <hip/hip_runtime.h>
#include <math.h>

#define DT_STEP 0.01
#define KL_W 0.01f
#define SMAX 2048
#define NEVMAX 512
#define HMAX 512
#define DMAX 128

typedef __attribute__((ext_vector_type(16))) _Float16 v16h;
typedef __attribute__((ext_vector_type(8)))  _Float16 v8h;
typedef __attribute__((ext_vector_type(8)))  float    v8f;

// ---------------- WMMA fragment helpers (A/B in this code's contiguous K order, wave32) --------
// A 16x32 f16: lane m = lane&15; halves e -> K = e + 16*(lane>>4).
// B 32x16 f16: lane n = lane&15; halves e -> K = e + 16*(lane>>4).
// C/D 16x16 f32: lane n = lane&15; vgpr vi -> row = vi + 8*(lane>>4).

__device__ __forceinline__ v16h load_a_f16(const _Float16* __restrict__ src, int ld,
                                           int row0, int k0, int maxrow) {
    int lane = threadIdx.x & 31;
    int r = row0 + (lane & 15);
    if (r >= maxrow) r = maxrow - 1;            // clamp: bad rows are masked at C store
    int kb = (lane >> 4) << 4;                  // 0 or 16
    const _Float16* p = src + (size_t)r * ld + k0 + kb;
    return *(const v16h*)(p);                   // K = kb .. kb+15 (32B contiguous)
}

// Bt is the transposed weight, f16, [N][K] row-major.
__device__ __forceinline__ v16h load_b_f16(const _Float16* __restrict__ Bt, int K,
                                           int k0, int n0) {
    int lane = threadIdx.x & 31;
    int n = lane & 15;
    int kb = (lane >> 4) << 4;                  // 0 or 16
    const _Float16* p = Bt + (size_t)(n0 + n) * K + k0 + kb;
    return *(const v16h*)(p);                   // 32B contiguous
}

__device__ __forceinline__ v8f wmma_f16(v8f c, v16h a, v16h b) {
    return __builtin_amdgcn_wmma_f32_16x16x32_f16(false, a, false, b, (short)0, c,
                                                  false, false);
}

__device__ __forceinline__ float softplusf(float x) {
    if (x > 20.0f) return x;
    return log1pf(__expf(x));
}

// ---------------- Operand prep: f16 copies / transposes ---------------------

__global__ void f32_to_f16_kernel(const float* __restrict__ src, int n,
                                  _Float16* __restrict__ dst) {
    int i = blockIdx.x * blockDim.x + threadIdx.x;
    if (i < n) dst[i] = (_Float16)src[i];
}

// src [R][C] f32 row-major -> dst [C][R] f16 row-major (B^T for WMMA B loads)
__global__ void transpose_to_f16_kernel(const float* __restrict__ src, int R, int C,
                                        _Float16* __restrict__ dst) {
    int i = blockIdx.x * blockDim.x + threadIdx.x;
    if (i < R * C) {
        int r = i / C, c = i % C;
        dst[(size_t)c * R + r] = (_Float16)src[i];
    }
}

// ---------------- Kernel 1: event schedule (sequential, replicates numpy) ---

__global__ void sched_kernel(const float* __restrict__ ev, int n_ev, const int* __restrict__ Tp,
                             int* __restrict__ ints, float* __restrict__ dt_s,
                             float* __restrict__ m_s, float* __restrict__ t0_s,
                             int* __restrict__ eidx_s, float* __restrict__ times,
                             int* __restrict__ proc) {
    if (threadIdx.x != 0 || blockIdx.x != 0) return;
    double T = (double)(*Tp);
    double t = 0.0;
    int idx = 0, s = 0, ne = 0;
    times[0] = 0.0f;
    while (t < T && s < SMAX) {
        double nt = t + DT_STEP;
        if (idx < n_ev && (double)ev[idx] <= nt) {
            double et = (double)ev[idx];
            double dd = et - t; if (dd < 0.0) dd = 0.0;
            dt_s[s] = (float)dd; m_s[s] = 1.0f; t0_s[s] = (float)t; eidx_s[s] = idx;
            if (ne < NEVMAX) proc[ne] = idx;
            ne++; t = et; idx++;
        } else {
            dt_s[s] = (float)DT_STEP; m_s[s] = 0.0f; t0_s[s] = (float)t; eidx_s[s] = 0;
            t = nt;
        }
        times[s + 1] = (float)t;
        s++;
    }
    ints[0] = s;        // S
    ints[1] = ne;       // n events
    ints[2] = s + 1;    // S+1
}

// ---------------- Kernel 2: WMMA GEMM  out[M,N] = A[M,K] @ B[K,N] + bias ----

__global__ void gemm_bias_kernel(const _Float16* __restrict__ A, int M, int K,
                                 const _Float16* __restrict__ Bt, int N,
                                 const float* __restrict__ bias,
                                 float* __restrict__ out) {
    int tilesM = (M + 15) >> 4;
    int tilesN = N >> 4;
    int lane = threadIdx.x & 31;
    int n_l = lane & 15;
    int rbase = (lane >> 4) << 3;
    for (int t = blockIdx.x; t < tilesM * tilesN; t += gridDim.x) {
        int row0 = (t / tilesN) * 16;
        int n0 = (t % tilesN) * 16;
        v8f c = {};
        for (int k0 = 0; k0 < K; k0 += 32) {
            v16h a = load_a_f16(A, K, row0, k0, M);
            v16h b = load_b_f16(Bt, K, k0, n0);
            c = wmma_f16(c, a, b);
        }
        float bb = bias[n0 + n_l];
#pragma unroll
        for (int vi = 0; vi < 8; ++vi) {
            int r = row0 + rbase + vi;
            if (r < M) out[(size_t)r * N + n0 + n_l] = c[vi] + bb;
        }
    }
}

// ---------------- Kernel 3: sequential Euler + jump scan --------------------

__global__ void scan_kernel(const int* __restrict__ ints,
                            const float* __restrict__ dt_s, const float* __restrict__ m_s,
                            const float* __restrict__ t0_s, const int* __restrict__ eidx_s,
                            const float* __restrict__ marks_proj,
                            const float* __restrict__ Wd1, const float* __restrict__ bd1,
                            const float* __restrict__ Wd2, const float* __restrict__ bd2,
                            const float* __restrict__ Wj1, const float* __restrict__ bj1,
                            const float* __restrict__ Wj2, const float* __restrict__ bj2,
                            const float* __restrict__ jsPtr, const float* __restrict__ z0,
                            int d, int h,
                            float* __restrict__ z_traj, float* __restrict__ z_ev,
                            _Float16* __restrict__ z_traj_h, _Float16* __restrict__ z_ev_h) {
    __shared__ float zt[DMAX + 1];      // [z, t0]
    __shared__ float z1s[DMAX];
    __shared__ float hs[HMAX];
    __shared__ float cat[2 * DMAX];
    __shared__ int evcnt_s;
    int tid = threadIdx.x;
    int S = ints[0];
    float js = *jsPtr;
    if (tid < d) {
        zt[tid] = z0[tid];
        z_traj[tid] = z0[tid];
        z_traj_h[tid] = (_Float16)z0[tid];
    }
    if (tid == 0) evcnt_s = 0;
    __syncthreads();
    for (int s = 0; s < S; ++s) {
        float dt = dt_s[s];
        float mi = m_s[s];
        if (tid == 0) zt[d] = t0_s[s];
        __syncthreads();
        // drift hidden: tanh(zt @ Wd1 + bd1)
        if (tid < h) {
            float acc = bd1[tid];
            for (int k = 0; k <= d; ++k) acc += zt[k] * Wd1[(size_t)k * h + tid];
            hs[tid] = tanhf(acc);
        }
        __syncthreads();
        if (tid < d) {
            float acc = bd2[tid];
            for (int k = 0; k < h; ++k) acc += hs[k] * Wd2[(size_t)k * d + tid];
            z1s[tid] = zt[tid] + acc * dt;
        }
        __syncthreads();
        int ec = evcnt_s;
        if (mi > 0.5f) {                       // uniform across block
            int ei = eidx_s[s];
            if (tid < d) {
                float z1 = z1s[tid];
                if (ec < NEVMAX) {
                    z_ev[(size_t)ec * d + tid] = z1;             // pre-jump latent
                    z_ev_h[(size_t)ec * d + tid] = (_Float16)z1;
                }
                cat[tid] = z1;
                cat[d + tid] = marks_proj[(size_t)ei * d + tid];
            }
            __syncthreads();
            if (tid < h) {
                float acc = bj1[tid];
                for (int k = 0; k < 2 * d; ++k) acc += cat[k] * Wj1[(size_t)k * h + tid];
                hs[tid] = acc > 0.0f ? acc : 0.0f;
            }
            __syncthreads();
            if (tid < d) {
                float acc = bj2[tid];
                for (int k = 0; k < h; ++k) acc += hs[k] * Wj2[(size_t)k * d + tid];
                float z2 = z1s[tid] + js * tanhf(acc);
                zt[tid] = z2;
                z_traj[(size_t)(s + 1) * d + tid] = z2;
                z_traj_h[(size_t)(s + 1) * d + tid] = (_Float16)z2;
            }
            if (tid == 0) evcnt_s = ec + 1;
            __syncthreads();
        } else {
            if (tid < d) {
                float z1 = z1s[tid];
                zt[tid] = z1;
                z_traj[(size_t)(s + 1) * d + tid] = z1;
                z_traj_h[(size_t)(s + 1) * d + tid] = (_Float16)z1;
            }
            __syncthreads();
        }
    }
}

// ---------------- Kernel 4: fused vol-net + intensity, WMMA, 16 rows/wave ---

__global__ void lam_kernel(const _Float16* __restrict__ Z, const int* __restrict__ cntPtr,
                           int d, int h,
                           const _Float16* __restrict__ WTv1, const float* __restrict__ bv1,
                           const _Float16* __restrict__ WTv2, const float* __restrict__ bv2,
                           const _Float16* __restrict__ WTiz, const float* __restrict__ biz,
                           const float* __restrict__ wiz2,
                           const _Float16* __restrict__ WTiv, const float* __restrict__ biv,
                           const float* __restrict__ wiv2,
                           const float* __restrict__ ibPtr, float* __restrict__ lamOut) {
    __shared__ __align__(32) _Float16 hbuf[16 * HMAX];   // relu hidden of vol net (f16)
    __shared__ __align__(32) _Float16 vbuf[16 * DMAX];   // volatility (f16)
    __shared__ float red[16 * 16];
    int count = *cntPtr;
    float ib = *ibPtr;
    int lane = threadIdx.x & 31;
    int n_l = lane & 15;
    int rbase = (lane >> 4) << 3;
    int tiles = (count + 15) >> 4;
    for (int t = blockIdx.x; t < tiles; t += gridDim.x) {
        int row0 = t * 16;
        // stage 1: hbuf = relu(Z @ Wv1 + bv1)   [16 x h]
        for (int n0 = 0; n0 < h; n0 += 16) {
            v8f c = {};
            for (int k0 = 0; k0 < d; k0 += 32) {
                v16h a = load_a_f16(Z, d, row0, k0, count);
                v16h b = load_b_f16(WTv1, d, k0, n0);
                c = wmma_f16(c, a, b);
            }
            float bb = bv1[n0 + n_l];
#pragma unroll
            for (int vi = 0; vi < 8; ++vi) {
                float x = c[vi] + bb;
                hbuf[(rbase + vi) * HMAX + n0 + n_l] = (_Float16)(x > 0.0f ? x : 0.0f);
            }
        }
        __syncthreads();
        // stage 2: vbuf = softplus(hbuf @ Wv2 + bv2)   [16 x d]
        for (int n0 = 0; n0 < d; n0 += 16) {
            v8f c = {};
            for (int k0 = 0; k0 < h; k0 += 32) {
                v16h a = load_a_f16(hbuf, HMAX, 0, k0, 16);
                v16h b = load_b_f16(WTv2, h, k0, n0);
                c = wmma_f16(c, a, b);
            }
            float bb = bv2[n0 + n_l];
#pragma unroll
            for (int vi = 0; vi < 8; ++vi)
                vbuf[(rbase + vi) * DMAX + n0 + n_l] = (_Float16)softplusf(c[vi] + bb);
        }
        __syncthreads();
        // stage 3: lam = softplus(relu(Z@Wiz+biz)@wiz2 + relu(v@Wiv+biv)@wiv2 + ib)
        float acc[8];
#pragma unroll
        for (int vi = 0; vi < 8; ++vi) acc[vi] = 0.0f;
        for (int n0 = 0; n0 < h; n0 += 16) {           // z channel
            v8f c = {};
            for (int k0 = 0; k0 < d; k0 += 32) {
                v16h a = load_a_f16(Z, d, row0, k0, count);
                v16h b = load_b_f16(WTiz, d, k0, n0);
                c = wmma_f16(c, a, b);
            }
            float bb = biz[n0 + n_l];
            float w2 = wiz2[n0 + n_l];
#pragma unroll
            for (int vi = 0; vi < 8; ++vi) {
                float x = c[vi] + bb;
                acc[vi] += (x > 0.0f ? x : 0.0f) * w2;
            }
        }
        for (int n0 = 0; n0 < h; n0 += 16) {           // vol channel
            v8f c = {};
            for (int k0 = 0; k0 < d; k0 += 32) {
                v16h a = load_a_f16(vbuf, DMAX, 0, k0, 16);
                v16h b = load_b_f16(WTiv, d, k0, n0);
                c = wmma_f16(c, a, b);
            }
            float bb = biv[n0 + n_l];
            float w2 = wiv2[n0 + n_l];
#pragma unroll
            for (int vi = 0; vi < 8; ++vi) {
                float x = c[vi] + bb;
                acc[vi] += (x > 0.0f ? x : 0.0f) * w2;
            }
        }
#pragma unroll
        for (int vi = 0; vi < 8; ++vi) red[(rbase + vi) * 16 + n_l] = acc[vi];
        __syncthreads();
        if (threadIdx.x < 16) {
            int r = row0 + (int)threadIdx.x;
            if (r < count) {
                float s = 0.0f;
                for (int n = 0; n < 16; ++n) s += red[threadIdx.x * 16 + n];
                lamOut[r] = softplusf(s + ib);
            }
        }
        __syncthreads();
    }
}

// ---------------- Kernel 5: reductions + loss -------------------------------

__device__ float block_reduce_sum(float v, float* sred) {
    int tid = threadIdx.x;
    sred[tid] = v;
    __syncthreads();
    for (int off = 128; off > 0; off >>= 1) {
        if (tid < off) sred[tid] += sred[tid + off];
        __syncthreads();
    }
    float r = sred[0];
    __syncthreads();
    return r;
}

__global__ void finalize_kernel(const int* __restrict__ ints,
                                const float* __restrict__ lam_traj, const float* __restrict__ lam_ev,
                                const float* __restrict__ times,
                                const float* __restrict__ z_traj, const float* __restrict__ z_ev,
                                const float* __restrict__ marks_proj, const int* __restrict__ proc,
                                const float* __restrict__ Wdec, const float* __restrict__ bdec,
                                int d, float* __restrict__ out) {
    __shared__ float sred[256];
    int tid = threadIdx.x;
    int S = ints[0];
    int ne = ints[1];

    float p = 0.0f;                                      // compensator integral
    for (int s = tid; s < S; s += 256) p += lam_traj[s] * (times[s + 1] - times[s]);
    float integral = block_reduce_sum(p, sred);

    p = 0.0f;                                            // log intensity at events
    for (int i = tid; i < ne; i += 256) p += logf(lam_ev[i] + 1e-8f);
    float log_int = block_reduce_sum(p, sred);

    p = 0.0f;                                            // KL
    size_t tot = (size_t)(S + 1) * (size_t)d;
    for (size_t i = tid; i < tot; i += 256) { float z = z_traj[i]; p += z * z; }
    float kl = KL_W * block_reduce_sum(p, sred) / (float)tot;

    p = 0.0f;                                            // Gaussian recon
    for (int i = 0; i < ne; ++i) {
        const float* zr = z_ev + (size_t)i * d;
        const float* mk = marks_proj + (size_t)proc[i] * d;
        for (int c = tid; c < d; c += 256) {
            float mean = bdec[c];
            for (int k = 0; k < d; ++k) mean += zr[k] * Wdec[(size_t)k * d + c];
            float diff = mk[c] - mean;
            p += diff * diff;
        }
    }
    float sq = block_reduce_sum(p, sred);
    float recon = -0.5f * sq - 0.5f * (float)d * 1.8378770664093453f * (float)ne;

    if (tid == 0) {
        float loss = -(log_int - integral + recon - kl);
        out[0] = loss; out[1] = log_int; out[2] = integral; out[3] = recon; out[4] = kl;
    }
}

// ---------------- Host launcher ---------------------------------------------

extern "C" void kernel_launch(void* const* d_in, const int* in_sizes, int n_in,
                              void* d_out, int out_size, void* d_ws, size_t ws_size,
                              hipStream_t stream) {
    const float* ev    = (const float*)d_in[0];
    const float* marks = (const float*)d_in[1];
    const int*   Tp    = (const int*)d_in[2];
    const float* Wp  = (const float*)d_in[3];  const float* bp  = (const float*)d_in[4];
    const float* Wd1 = (const float*)d_in[5];  const float* bd1 = (const float*)d_in[6];
    const float* Wd2 = (const float*)d_in[7];  const float* bd2 = (const float*)d_in[8];
    const float* Wv1 = (const float*)d_in[9];  const float* bv1 = (const float*)d_in[10];
    const float* Wv2 = (const float*)d_in[11]; const float* bv2 = (const float*)d_in[12];
    const float* Wj1 = (const float*)d_in[13]; const float* bj1 = (const float*)d_in[14];
    const float* Wj2 = (const float*)d_in[15]; const float* bj2 = (const float*)d_in[16];
    const float* js  = (const float*)d_in[17];
    const float* z0  = (const float*)d_in[18];
    const float* Wiz = (const float*)d_in[19]; const float* biz = (const float*)d_in[20];
    const float* wiz2= (const float*)d_in[21];
    const float* Wiv = (const float*)d_in[22]; const float* biv = (const float*)d_in[23];
    const float* wiv2= (const float*)d_in[24];
    const float* ib  = (const float*)d_in[25];
    const float* Wdec= (const float*)d_in[26]; const float* bdec= (const float*)d_in[27];

    int n_ev = in_sizes[0];
    int d    = in_sizes[18];
    int h    = in_sizes[6];
    int d_in_dim = in_sizes[1] / n_ev;

    // carve workspace
    char* base = (char*)d_ws;
    size_t off = 0;
    auto carve = [&](size_t bytes) -> void* {
        void* p = base + off;
        off += bytes;
        off = (off + 255) & ~(size_t)255;
        return p;
    };
    int*   ints       = (int*)carve(64);
    float* dt_s       = (float*)carve((size_t)SMAX * 4);
    float* m_s        = (float*)carve((size_t)SMAX * 4);
    float* t0_s       = (float*)carve((size_t)SMAX * 4);
    float* times      = (float*)carve((size_t)(SMAX + 1) * 4);
    int*   eidx_s     = (int*)carve((size_t)SMAX * 4);
    int*   proc       = (int*)carve((size_t)NEVMAX * 4);
    float* marks_proj = (float*)carve((size_t)n_ev * d * 4);
    float* z_traj     = (float*)carve((size_t)(SMAX + 1) * d * 4);
    float* z_ev       = (float*)carve((size_t)NEVMAX * d * 4);
    float* lam_traj   = (float*)carve((size_t)(SMAX + 1) * 4);
    float* lam_ev     = (float*)carve((size_t)NEVMAX * 4);
    _Float16* marks_h = (_Float16*)carve((size_t)n_ev * d_in_dim * 2);
    _Float16* WTp     = (_Float16*)carve((size_t)d * d_in_dim * 2);       // [d][d_in]
    _Float16* WTv1    = (_Float16*)carve((size_t)h * d * 2);              // [h][d]
    _Float16* WTv2    = (_Float16*)carve((size_t)d * h * 2);              // [d][h]
    _Float16* WTiz    = (_Float16*)carve((size_t)h * d * 2);              // [h][d]
    _Float16* WTiv    = (_Float16*)carve((size_t)h * d * 2);              // [h][d]
    _Float16* z_traj_h= (_Float16*)carve((size_t)(SMAX + 1) * d * 2);
    _Float16* z_ev_h  = (_Float16*)carve((size_t)NEVMAX * d * 2);
    (void)ws_size; (void)n_in; (void)out_size;

    // 0) operand prep: f16 copies and B^T weights
    {
        int n = n_ev * d_in_dim;
        f32_to_f16_kernel<<<(n + 255) / 256, 256, 0, stream>>>(marks, n, marks_h);
        transpose_to_f16_kernel<<<(d_in_dim * d + 255) / 256, 256, 0, stream>>>(Wp, d_in_dim, d, WTp);
        transpose_to_f16_kernel<<<(d * h + 255) / 256, 256, 0, stream>>>(Wv1, d, h, WTv1);
        transpose_to_f16_kernel<<<(h * d + 255) / 256, 256, 0, stream>>>(Wv2, h, d, WTv2);
        transpose_to_f16_kernel<<<(d * h + 255) / 256, 256, 0, stream>>>(Wiz, d, h, WTiz);
        transpose_to_f16_kernel<<<(d * h + 255) / 256, 256, 0, stream>>>(Wiv, d, h, WTiv);
    }

    // 1) schedule (sequential)
    sched_kernel<<<1, 1, 0, stream>>>(ev, n_ev, Tp, ints, dt_s, m_s, t0_s, eidx_s, times, proc);

    // 2) marks projection: WMMA GEMM
    int tilesProj = ((n_ev + 15) / 16) * (d / 16);
    gemm_bias_kernel<<<tilesProj, 32, 0, stream>>>(marks_h, n_ev, d_in_dim, WTp, d, bp, marks_proj);

    // 3) sequential SDE scan
    scan_kernel<<<1, 512, 0, stream>>>(ints, dt_s, m_s, t0_s, eidx_s, marks_proj,
                                       Wd1, bd1, Wd2, bd2, Wj1, bj1, Wj2, bj2,
                                       js, z0, d, h, z_traj, z_ev, z_traj_h, z_ev_h);

    // 4) trajectory intensities (count = S+1) and event intensities (count = ne)
    lam_kernel<<<(SMAX / 16) + 1, 32, 0, stream>>>(z_traj_h, ints + 2, d, h,
                                                   WTv1, bv1, WTv2, bv2,
                                                   WTiz, biz, wiz2, WTiv, biv, wiv2,
                                                   ib, lam_traj);
    lam_kernel<<<NEVMAX / 16, 32, 0, stream>>>(z_ev_h, ints + 1, d, h,
                                               WTv1, bv1, WTv2, bv2,
                                               WTiz, biz, wiz2, WTiv, biv, wiv2,
                                               ib, lam_ev);

    // 5) losses
    finalize_kernel<<<1, 256, 0, stream>>>(ints, lam_traj, lam_ev, times, z_traj, z_ev,
                                           marks_proj, proc, Wdec, bdec, d, (float*)d_out);
}
